// Model_34187939676854
// MI455X (gfx1250) — matrix operand order repaired and checked by reading, not compiled, not run on hardware
//
#include <hip/hip_runtime.h>
#include <hip/hip_bf16.h>
#include <math.h>

typedef __attribute__((ext_vector_type(16))) _Float16 v16h;
typedef __attribute__((ext_vector_type(8)))  _Float16 v8h;
typedef __attribute__((ext_vector_type(8)))  float    v8f;

#define DM    512
#define FFD   2048
#define NH    8
#define HE    64
#define BZ    16
#define SEQL  512
#define DECL  288
#define PREDL 192
#define CIN   21
#define COUT  21
#define MARKN 4

// ---------------------------------------------------------------------------
// WMMA GEMM: C[M,N] = scale * (A[M,K] @ Bt[N,K]^T) + bias + resid, opt. GELU.
// A, Bt are f16; Bt is stored [N,K] row-major so both operands use the same
// LDS fragment loader. Batched over z = zb*Hdim + zh with independent pointer
// strides (used for per-(batch,head) attention GEMMs).
// ---------------------------------------------------------------------------
#define TM 128
#define TN 128
#define TK 32
#define LDSS 40   // padded LDS row stride (halves) -> conflict-minimal b128 reads

__launch_bounds__(256)
__global__ void wmma_gemm_kernel(
    const _Float16* __restrict__ A, long long lda, long long sAb, long long sAh,
    const _Float16* __restrict__ Bt, long long ldb, long long sBb, long long sBh,
    float* Cf32, _Float16* Cf16, long long ldc, long long sCb, long long sCh,
    const float* __restrict__ bias, const float* __restrict__ resid,
    int M, int N, int K, int Hdim, float scale, int do_gelu)
{
  __shared__ _Float16 As[TM * LDSS];
  __shared__ _Float16 Bs[TN * LDSS];

  const int tid  = threadIdx.x;
  const int lane = tid & 31;
  const int wave = tid >> 5;
  const int wm   = wave >> 1;    // 0..3 : 32-row strip
  const int wn   = wave & 1;     // 0..1 : 64-col strip
  const int m0   = blockIdx.y * TM;
  const int n0   = blockIdx.x * TN;

  const int z  = blockIdx.z;
  const int zb = z / Hdim;
  const int zh = z % Hdim;
  A  += (long long)zb * sAb + (long long)zh * sAh;
  Bt += (long long)zb * sBb + (long long)zh * sBh;
  const long long coff = (long long)zb * sCb + (long long)zh * sCh;
  if (Cf32)  Cf32  += coff;
  if (Cf16)  Cf16  += coff;
  if (resid) resid += coff;

  v8f acc[2][4];
#pragma unroll
  for (int i = 0; i < 2; ++i)
#pragma unroll
    for (int j = 0; j < 4; ++j)
#pragma unroll
      for (int g = 0; g < 8; ++g) acc[i][j][g] = 0.0f;

  const int lrow = lane & 15;
  const int lhi  = lane >> 4;

  for (int k0 = 0; k0 < K; k0 += TK) {
    // Stage A (TM x TK) and Bt (TN x TK). 256 threads * 8 halves * 2 reps each.
#pragma unroll
    for (int rep = 0; rep < 2; ++rep) {
      const int linear = rep * 2048 + tid * 8;
      const int r = linear >> 5;
      const int c = linear & 31;
      {
        const int gr = m0 + r;
        v8h val;
#pragma unroll
        for (int q = 0; q < 8; ++q) val[q] = (_Float16)0.0f;
        if (gr < M) val = *(const v8h*)(A + (long long)gr * lda + (k0 + c));
        *(v8h*)(&As[r * LDSS + c]) = val;
      }
      {
        const int gn = n0 + r;
        v8h val;
#pragma unroll
        for (int q = 0; q < 8; ++q) val[q] = (_Float16)0.0f;
        if (gn < N) val = *(const v8h*)(Bt + (long long)gn * ldb + (k0 + c));
        *(v8h*)(&Bs[r * LDSS + c]) = val;
      }
    }
    __syncthreads();

    v16h afrag[2], bfrag[4];
#pragma unroll
    for (int i = 0; i < 2; ++i) {
      // A 16x32 f16 layout: lane<16 -> row=lane, K in {0..7,16..23};
      //                     lane>=16 -> row=lane-16, K in {8..15,24..31}.
      const _Float16* ap = &As[(wm * 32 + i * 16 + lrow) * LDSS + lhi * 8];
      v8h lo = *(const v8h*)ap;
      v8h hi = *(const v8h*)(ap + 16);
#pragma unroll
      for (int q = 0; q < 8; ++q) { afrag[i][q] = lo[q]; afrag[i][8 + q] = hi[q]; }
    }
#pragma unroll
    for (int j = 0; j < 4; ++j) {
      // B 32x16 f16 layout: lanes 0-15 -> N=lane, K in {0..7,16..23};
      //                     lanes 16-31 -> N=lane-16, K in {8..15,24..31}.
      const _Float16* bp = &Bs[(wn * 64 + j * 16 + lrow) * LDSS + lhi * 8];
      v8h lo = *(const v8h*)bp;
      v8h hi = *(const v8h*)(bp + 16);
#pragma unroll
      for (int q = 0; q < 8; ++q) { bfrag[j][q] = lo[q]; bfrag[j][8 + q] = hi[q]; }
    }
#pragma unroll
    for (int i = 0; i < 2; ++i)
#pragma unroll
      for (int j = 0; j < 4; ++j)
        acc[i][j] = __builtin_amdgcn_wmma_f32_16x16x32_f16(
            false, afrag[i], false, bfrag[j], (short)0, acc[i][j], false, false);
    __syncthreads();
  }

  // Epilogue. C/D layout: lanes 0-15 -> M = g, lanes 16-31 -> M = 8+g; N = lane&15.
#pragma unroll
  for (int i = 0; i < 2; ++i) {
#pragma unroll
    for (int j = 0; j < 4; ++j) {
      const int c = n0 + wn * 64 + j * 16 + lrow;
      if (c >= N) continue;
      const float bv = bias ? bias[c] : 0.0f;
#pragma unroll
      for (int g = 0; g < 8; ++g) {
        const int r = m0 + wm * 32 + i * 16 + lhi * 8 + g;
        if (r >= M) continue;
        float v = acc[i][j][g] * scale + bv;
        if (resid)   v += resid[(long long)r * ldc + c];
        if (do_gelu) v = 0.5f * v * (1.0f + erff(v * 0.70710678118654752f));
        if (Cf32) Cf32[(long long)r * ldc + c] = v;
        if (Cf16) Cf16[(long long)r * ldc + c] = (_Float16)v;
      }
    }
  }
}

// ---------------------------------------------------------------------------
// Row softmax over f32 scores -> f16 probs, with optional causal mask.
// One block per row (row = z*L + l), row length S.
// ---------------------------------------------------------------------------
__launch_bounds__(256)
__global__ void softmax_kernel(const float* __restrict__ scores,
                               _Float16* __restrict__ probs,
                               int L, int S, int causal)
{
  __shared__ float red[256];
  const long long row = blockIdx.x;
  const int l = (int)(row % L);
  const float* src = scores + row * (long long)S;
  _Float16* dst = probs + row * (long long)S;
  const int limit = causal ? (l + 1) : S;
  const int tid = threadIdx.x;

  float mx = -3.0e38f;
  for (int s = tid; s < limit; s += 256) mx = fmaxf(mx, src[s]);
  red[tid] = mx; __syncthreads();
  for (int off = 128; off > 0; off >>= 1) {
    if (tid < off) red[tid] = fmaxf(red[tid], red[tid + off]);
    __syncthreads();
  }
  mx = red[0]; __syncthreads();

  float sm = 0.0f;
  for (int s = tid; s < limit; s += 256) sm += __expf(src[s] - mx);
  red[tid] = sm; __syncthreads();
  for (int off = 128; off > 0; off >>= 1) {
    if (tid < off) red[tid] += red[tid + off];
    __syncthreads();
  }
  const float inv = 1.0f / red[0];

  for (int s = tid; s < S; s += 256) {
    const float p = (s < limit) ? __expf(src[s] - mx) * inv : 0.0f;
    dst[s] = (_Float16)p;
  }
}

// ---------------------------------------------------------------------------
// LayerNorm over D=512, one block (256 threads, 2 elems/thread) per token.
// Writes f32 stream and f16 copy (either may be null). Safe in-place.
// ---------------------------------------------------------------------------
__launch_bounds__(256)
__global__ void layernorm_kernel(const float* __restrict__ in,
                                 const float* __restrict__ gw,
                                 const float* __restrict__ bw,
                                 float* out32, _Float16* out16)
{
  __shared__ float red[256];
  const long long t = blockIdx.x;
  const float* x = in + t * DM;
  const int tid = threadIdx.x;
  const float a = x[tid], b = x[tid + 256];
  red[tid] = a + b; __syncthreads();
  for (int off = 128; off > 0; off >>= 1) {
    if (tid < off) red[tid] += red[tid + off];
    __syncthreads();
  }
  const float mean = red[0] * (1.0f / DM); __syncthreads();
  const float d0 = a - mean, d1 = b - mean;
  red[tid] = d0 * d0 + d1 * d1; __syncthreads();
  for (int off = 128; off > 0; off >>= 1) {
    if (tid < off) red[tid] += red[tid + off];
    __syncthreads();
  }
  const float rstd = rsqrtf(red[0] * (1.0f / DM) + 1e-5f);
  const float y0 = d0 * rstd * gw[tid] + bw[tid];
  const float y1 = d1 * rstd * gw[tid + 256] + bw[tid + 256];
  if (out32) { out32[t * DM + tid] = y0; out32[t * DM + tid + 256] = y1; }
  if (out16) { out16[t * DM + tid] = (_Float16)y0; out16[t * DM + tid + 256] = (_Float16)y1; }
}

// ---------------------------------------------------------------------------
// Embedding: circular conv1d(C->D,k=3) + x_mark @ temp_w + sinusoidal PE.
// One block per (b,t); window + marks staged in LDS.
// ---------------------------------------------------------------------------
__launch_bounds__(256)
__global__ void embed_kernel(const float* __restrict__ x,
                             const float* __restrict__ xmark,
                             const float* __restrict__ conv_w,
                             const float* __restrict__ temp_w,
                             float* __restrict__ out32,
                             _Float16* __restrict__ out16, int T)
{
  __shared__ float win[3][CIN];
  __shared__ float mk[MARKN];
  const int bt = blockIdx.x;
  const int b = bt / T, t = bt % T;
  const int tid = threadIdx.x;
  if (tid < 3 * CIN) {
    const int k = tid / CIN, c = tid % CIN;
    int tt = t - 1 + k; tt = (tt + T) % T;       // circular padding
    win[k][c] = x[((long long)b * T + tt) * CIN + c];
  }
  if (tid < MARKN) mk[tid] = xmark[((long long)b * T + t) * MARKN + tid];
  __syncthreads();
  const float nlog = -logf(10000.0f) / (float)DM;
  for (int d = tid; d < DM; d += 256) {
    float acc = 0.0f;
    const float* w = conv_w + (long long)d * (CIN * 3);
#pragma unroll
    for (int k = 0; k < 3; ++k)
      for (int c = 0; c < CIN; ++c)
        acc += win[k][c] * w[c * 3 + k];
#pragma unroll
    for (int m = 0; m < MARKN; ++m) acc += mk[m] * temp_w[m * DM + d];
    const float freq = __expf((float)(2 * (d >> 1)) * nlog);
    acc += (d & 1) ? __cosf((float)t * freq) : __sinf((float)t * freq);
    const long long o = ((long long)b * T + t) * DM + d;
    out32[o] = acc;
    out16[o] = (_Float16)acc;
  }
}

// Wt[n*Din + k] = (f16) W[k*Dout + n]   (W is [Din, Dout] f32)
__global__ void transpose_w_kernel(const float* __restrict__ W,
                                   _Float16* __restrict__ Wt, int Din, int Dout)
{
  const long long i = (long long)blockIdx.x * 256 + threadIdx.x;
  if (i >= (long long)Din * Dout) return;
  const int n = (int)(i / Din);
  const int k = (int)(i % Din);
  Wt[i] = (_Float16)W[(long long)k * Dout + n];
}

// v16 [B*S, D] -> vt [B, H, 64, S]  (so AV GEMM can use the [N,K] loader)
__global__ void repack_v_kernel(const _Float16* __restrict__ v,
                                _Float16* __restrict__ vt, int S, long long total)
{
  const long long i = (long long)blockIdx.x * 256 + threadIdx.x;
  if (i >= total) return;
  const int d = (int)(i % DM);
  const long long t = i / DM;
  const int s = (int)(t % S);
  const int b = (int)(t / S);
  const int h = d >> 6, e = d & 63;
  vt[(((long long)b * NH + h) * HE + e) * S + s] = v[i];
}

// d_out[b, p, c] = y[b, DECL-PREDL+p, c]
__global__ void slice_out_kernel(const float* __restrict__ y, float* __restrict__ out)
{
  const long long total = (long long)BZ * PREDL * COUT;
  const long long i = (long long)blockIdx.x * 256 + threadIdx.x;
  if (i >= total) return;
  const int c = (int)(i % COUT);
  const long long t = i / COUT;
  const int p = (int)(t % PREDL);
  const int b = (int)(t / PREDL);
  out[i] = y[((long long)b * DECL + (DECL - PREDL) + p) * COUT + c];
}

// ---------------------------------------------------------------------------
// Host orchestration
// ---------------------------------------------------------------------------
struct AttnIdx { int wq, bq, wk, bk, wv, bv, wo, bo; };
struct FFIdx   { int w1, b1, w2, b2; };
struct LNIdx   { int g, b; };

extern "C" void kernel_launch(void* const* d_in, const int* in_sizes, int n_in,
                              void* d_out, int out_size, void* d_ws, size_t ws_size,
                              hipStream_t stream)
{
  (void)in_sizes; (void)n_in; (void)out_size; (void)ws_size;

  // ---- input index map (setup_inputs() insertion order, depth-first) ----
  int ix = 0;
  const int I_XENC = ix++, I_XMENC = ix++, I_XDEC = ix++, I_XMDEC = ix++;
  const int I_ENC_CONV = ix++, I_ENC_TEMP = ix++;
  const int I_DEC_CONV = ix++, I_DEC_TEMP = ix++;
  auto takeAttn = [&]() { AttnIdx a; a.wq=ix++; a.bq=ix++; a.wk=ix++; a.bk=ix++;
                          a.wv=ix++; a.bv=ix++; a.wo=ix++; a.bo=ix++; return a; };
  auto takeFF   = [&]() { FFIdx f; f.w1=ix++; f.b1=ix++; f.w2=ix++; f.b2=ix++; return f; };
  auto takeLN   = [&]() { LNIdx l; l.g=ix++; l.b=ix++; return l; };
  AttnIdx featA[2]; FFIdx featF[2]; LNIdx featN[2];
  for (int i = 0; i < 2; ++i) { featA[i]=takeAttn(); featF[i]=takeFF(); featN[i]=takeLN(); }
  AttnIdx tmpA[2]; FFIdx tmpF[2]; LNIdx tmpN1[2], tmpN2[2];
  for (int i = 0; i < 2; ++i) { tmpA[i]=takeAttn(); tmpF[i]=takeFF(); tmpN1[i]=takeLN(); tmpN2[i]=takeLN(); }
  LNIdx tempNorm = takeLN();
  AttnIdx decS[2], decC[2]; FFIdx decF[2]; LNIdx decN1[2], decN2[2], decN3[2];
  for (int i = 0; i < 2; ++i) { decS[i]=takeAttn(); decC[i]=takeAttn(); decF[i]=takeFF();
                                decN1[i]=takeLN(); decN2[i]=takeLN(); decN3[i]=takeLN(); }
  LNIdx decNorm = takeLN();
  const int I_PROJ_W = ix++, I_PROJ_B = ix++;

  auto F32 = [&](int i) { return (const float*)d_in[i]; };

  // ---- workspace bump allocator ----
  char* base = (char*)d_ws;
  size_t off = 0;
  auto alloc = [&](size_t bytes) -> void* {
    void* p = base + off;
    off = (off + bytes + 255) & ~(size_t)255;
    return p;
  };

  const int TOKE = BZ * SEQL;   // 8192 encoder tokens
  const int TOKD = BZ * DECL;   // 4608 decoder tokens

  float*    x32   = (float*)   alloc((size_t)TOKE * DM * 4);
  _Float16* x16   = (_Float16*)alloc((size_t)TOKE * DM * 2);
  float*    y32   = (float*)   alloc((size_t)TOKD * DM * 4);
  _Float16* y16   = (_Float16*)alloc((size_t)TOKD * DM * 2);
  float*    tmp32 = (float*)   alloc((size_t)TOKE * DM * 4);
  _Float16* q16   = (_Float16*)alloc((size_t)TOKE * DM * 2);
  _Float16* k16   = (_Float16*)alloc((size_t)TOKE * DM * 2);
  _Float16* v16   = (_Float16*)alloc((size_t)TOKE * DM * 2);
  _Float16* vt16  = (_Float16*)alloc((size_t)TOKE * DM * 2);
  _Float16* ctx16 = (_Float16*)alloc((size_t)TOKE * DM * 2);
  _Float16* hid16 = (_Float16*)alloc((size_t)TOKE * FFD * 2);
  float*    scores= (float*)   alloc((size_t)BZ * NH * SEQL * SEQL * 4);
  _Float16* probs = (_Float16*)alloc((size_t)BZ * NH * SEQL * SEQL * 2);
  float*    pout  = (float*)   alloc((size_t)TOKD * COUT * 4);

  // ---- on-device weight transpose + f32->f16 (stateless, graph-safe) ----
  auto prepW = [&](int i, int Din, int Dout) -> const _Float16* {
    _Float16* dst = (_Float16*)alloc((size_t)Din * Dout * sizeof(_Float16));
    const long long tot = (long long)Din * Dout;
    transpose_w_kernel<<<(unsigned)((tot + 255) / 256), 256, 0, stream>>>(F32(i), dst, Din, Dout);
    return dst;
  };
  const _Float16 *featWv[2], *featWo[2], *featW1[2], *featW2[2];
  for (int i = 0; i < 2; ++i) {
    featWv[i] = prepW(featA[i].wv, DM, DM);  featWo[i] = prepW(featA[i].wo, DM, DM);
    featW1[i] = prepW(featF[i].w1, DM, FFD); featW2[i] = prepW(featF[i].w2, FFD, DM);
  }
  const _Float16 *tWq[2], *tWk[2], *tWv[2], *tWo[2], *tW1[2], *tW2[2];
  for (int i = 0; i < 2; ++i) {
    tWq[i] = prepW(tmpA[i].wq, DM, DM); tWk[i] = prepW(tmpA[i].wk, DM, DM);
    tWv[i] = prepW(tmpA[i].wv, DM, DM); tWo[i] = prepW(tmpA[i].wo, DM, DM);
    tW1[i] = prepW(tmpF[i].w1, DM, FFD); tW2[i] = prepW(tmpF[i].w2, FFD, DM);
  }
  const _Float16 *sWq[2], *sWk[2], *sWv[2], *sWo[2];
  const _Float16 *cWq[2], *cWk[2], *cWv[2], *cWo[2];
  const _Float16 *dW1[2], *dW2[2];
  for (int i = 0; i < 2; ++i) {
    sWq[i] = prepW(decS[i].wq, DM, DM); sWk[i] = prepW(decS[i].wk, DM, DM);
    sWv[i] = prepW(decS[i].wv, DM, DM); sWo[i] = prepW(decS[i].wo, DM, DM);
    cWq[i] = prepW(decC[i].wq, DM, DM); cWk[i] = prepW(decC[i].wk, DM, DM);
    cWv[i] = prepW(decC[i].wv, DM, DM); cWo[i] = prepW(decC[i].wo, DM, DM);
    dW1[i] = prepW(decF[i].w1, DM, FFD); dW2[i] = prepW(decF[i].w2, FFD, DM);
  }
  const _Float16* projT = prepW(I_PROJ_W, DM, COUT);

  // ---- helpers ----
  auto gemm = [&](const _Float16* A, long long lda, long long sAb, long long sAh,
                  const _Float16* Bt, long long ldb, long long sBb, long long sBh,
                  float* Cf, _Float16* Ch, long long ldc, long long sCb, long long sCh,
                  const float* bias, const float* resid,
                  int M, int N, int K, int Zb, int Zh, float scale, int gelu) {
    dim3 grid((N + TN - 1) / TN, (M + TM - 1) / TM, Zb * Zh);
    wmma_gemm_kernel<<<grid, 256, 0, stream>>>(A, lda, sAb, sAh, Bt, ldb, sBb, sBh,
        Cf, Ch, ldc, sCb, sCh, bias, resid, M, N, K, Zh, scale, gelu);
  };
  auto lnorm = [&](const float* in, LNIdx ln, float* o32, _Float16* o16, int tokens) {
    layernorm_kernel<<<tokens, 256, 0, stream>>>(in, F32(ln.g), F32(ln.b), o32, o16);
  };
  auto attention = [&](const _Float16* xq, int Lq, const _Float16* xkv, int Skv,
                       AttnIdx a, const _Float16* wqT, const _Float16* wkT,
                       const _Float16* wvT, const _Float16* woT,
                       bool causal, const float* resid32, float* out32) {
    const int Mq = BZ * Lq, Mkv = BZ * Skv;
    gemm(xq,  DM,0,0, wqT, DM,0,0, nullptr, q16, DM,0,0, F32(a.bq), nullptr, Mq,  DM, DM, 1,1, 1.f, 0);
    gemm(xkv, DM,0,0, wkT, DM,0,0, nullptr, k16, DM,0,0, F32(a.bk), nullptr, Mkv, DM, DM, 1,1, 1.f, 0);
    gemm(xkv, DM,0,0, wvT, DM,0,0, nullptr, v16, DM,0,0, F32(a.bv), nullptr, Mkv, DM, DM, 1,1, 1.f, 0);
    const long long totv = (long long)Mkv * DM;
    repack_v_kernel<<<(unsigned)((totv + 255) / 256), 256, 0, stream>>>(v16, vt16, Skv, totv);
    // scores[z][l][s] = Q_h @ K_h^T / sqrt(64)
    gemm(q16, DM, (long long)Lq * DM, HE,
         k16, DM, (long long)Skv * DM, HE,
         scores, nullptr, Skv, (long long)NH * Lq * Skv, (long long)Lq * Skv,
         nullptr, nullptr, Lq, Skv, HE, BZ, NH, 0.125f, 0);
    softmax_kernel<<<(unsigned)(BZ * NH * Lq), 256, 0, stream>>>(scores, probs, Lq, Skv, causal ? 1 : 0);
    // ctx[(b*Lq+l)*DM + h*64 + e] = probs @ V
    gemm(probs, Skv, (long long)NH * Lq * Skv, (long long)Lq * Skv,
         vt16,  Skv, (long long)NH * HE * Skv, (long long)HE * Skv,
         nullptr, ctx16, DM, (long long)Lq * DM, HE,
         nullptr, nullptr, Lq, HE, Skv, BZ, NH, 1.f, 0);
    gemm(ctx16, DM,0,0, woT, DM,0,0, out32, nullptr, DM,0,0, F32(a.bo), resid32,
         Mq, DM, DM, 1,1, 1.f, 0);
  };
  auto ffn_block = [&](float* r32, const _Float16* in16, FFIdx f,
                       const _Float16* w1T, const _Float16* w2T, int tokens) {
    gemm(in16, DM,0,0, w1T, DM,0,0, nullptr, hid16, FFD,0,0, F32(f.b1), nullptr,
         tokens, FFD, DM, 1,1, 1.f, 1 /*gelu*/);
    gemm(hid16, FFD,0,0, w2T, FFD,0,0, tmp32, nullptr, DM,0,0, F32(f.b2), r32,
         tokens, DM, FFD, 1,1, 1.f, 0);
  };

  // ================= encoder =================
  embed_kernel<<<BZ * SEQL, 256, 0, stream>>>(F32(I_XENC), F32(I_XMENC),
      F32(I_ENC_CONV), F32(I_ENC_TEMP), x32, x16, SEQL);

  // Feature blocks: attn degenerates to Wo(Wv(x)); same LN applied twice.
  for (int i = 0; i < 2; ++i) {
    gemm(x16, DM,0,0, featWv[i], DM,0,0, nullptr, q16, DM,0,0, F32(featA[i].bv), nullptr,
         TOKE, DM, DM, 1,1, 1.f, 0);
    gemm(q16, DM,0,0, featWo[i], DM,0,0, tmp32, nullptr, DM,0,0, F32(featA[i].bo), x32,
         TOKE, DM, DM, 1,1, 1.f, 0);
    lnorm(tmp32, featN[i], x32, x16, TOKE);
    ffn_block(x32, x16, featF[i], featW1[i], featW2[i], TOKE);
    lnorm(tmp32, featN[i], x32, x16, TOKE);
  }
  // Temporal encoder layers
  for (int i = 0; i < 2; ++i) {
    attention(x16, SEQL, x16, SEQL, tmpA[i], tWq[i], tWk[i], tWv[i], tWo[i],
              false, x32, tmp32);
    lnorm(tmp32, tmpN1[i], x32, x16, TOKE);
    ffn_block(x32, x16, tmpF[i], tW1[i], tW2[i], TOKE);
    lnorm(tmp32, tmpN2[i], x32, x16, TOKE);
  }
  lnorm(x32, tempNorm, x32, x16, TOKE);   // x16 = encoder output (cross K/V src)

  // ================= decoder =================
  embed_kernel<<<BZ * DECL, 256, 0, stream>>>(F32(I_XDEC), F32(I_XMDEC),
      F32(I_DEC_CONV), F32(I_DEC_TEMP), y32, y16, DECL);
  for (int i = 0; i < 2; ++i) {
    attention(y16, DECL, y16, DECL, decS[i], sWq[i], sWk[i], sWv[i], sWo[i],
              true, y32, tmp32);
    lnorm(tmp32, decN1[i], y32, y16, TOKD);
    attention(y16, DECL, x16, SEQL, decC[i], cWq[i], cWk[i], cWv[i], cWo[i],
              false, y32, tmp32);
    lnorm(tmp32, decN2[i], y32, y16, TOKD);
    ffn_block(y32, y16, decF[i], dW1[i], dW2[i], TOKD);
    lnorm(tmp32, decN3[i], y32, y16, TOKD);
  }
  lnorm(y32, decNorm, y32, y16, TOKD);

  // projection + slice last PRED_LEN steps
  gemm(y16, DM,0,0, projT, DM,0,0, pout, nullptr, COUT,0,0, F32(I_PROJ_B), nullptr,
       TOKD, COUT, DM, 1,1, 1.f, 0);
  const long long tot = (long long)BZ * PREDL * COUT;
  slice_out_kernel<<<(unsigned)((tot + 255) / 256), 256, 0, stream>>>(pout, (float*)d_out);
}
